// FullGraphTransformerBlock_46832323396449
// MI455X (gfx1250) — matrix operand order repaired and checked by reading, not compiled, not run on hardware
//
#include <hip/hip_runtime.h>

typedef __attribute__((ext_vector_type(16))) __bf16 v16bf;
typedef __attribute__((ext_vector_type(8)))  float  v8f;

#define HID 128
#define NH  4
#define CH  32

// ---------------------------------------------------------------------------
// helpers
// ---------------------------------------------------------------------------
__device__ __forceinline__ float blk_reduce_sum128(float v, float* sm) {
    int t = threadIdx.x;
    sm[t] = v;
    __syncthreads();
#pragma unroll
    for (int s = 64; s > 0; s >>= 1) {
        if (t < s) sm[t] += sm[t + s];
        __syncthreads();
    }
    float r = sm[0];
    __syncthreads();
    return r;
}

__device__ __forceinline__ void atomicMaxF(float* addr, float v) {
    // IEEE ordered atomic max (works with init bits 0xFF800000 == -inf)
    if (v >= 0.0f) atomicMax((int*)addr, __float_as_int(v));
    else           atomicMin((unsigned int*)addr, __float_as_uint(v));
}

// ---------------------------------------------------------------------------
// f32 -> bf16 conversion
// ---------------------------------------------------------------------------
__global__ void k_f2bf(const float* __restrict__ s, __bf16* __restrict__ d, int n) {
    int i = blockIdx.x * blockDim.x + threadIdx.x;
    if (i < n) d[i] = (__bf16)s[i];
}

// ---------------------------------------------------------------------------
// init: out accumulator = 0, m = -inf, denom = 0
// ---------------------------------------------------------------------------
__global__ void k_init(float* __restrict__ outAcc, unsigned int* __restrict__ m,
                       float* __restrict__ denom, int n_nodes) {
    int i = blockIdx.x * blockDim.x + threadIdx.x;
    int total = n_nodes * HID;
    if (i < total) outAcc[i] = 0.0f;
    if (i < n_nodes * NH) { m[i] = 0xFF800000u; denom[i] = 0.0f; }
}

// ---------------------------------------------------------------------------
// LayerNorm (128 ch) -> bf16   (one block of 128 threads per node)
// ---------------------------------------------------------------------------
__global__ void k_ln_bf16(const float* __restrict__ x, const float* __restrict__ g,
                          const float* __restrict__ b, __bf16* __restrict__ out) {
    __shared__ float sm[HID];
    int n = blockIdx.x, c = threadIdx.x;
    float xv  = x[(size_t)n * HID + c];
    float mu  = blk_reduce_sum128(xv, sm) * (1.0f / HID);
    float d   = xv - mu;
    float var = blk_reduce_sum128(d * d, sm) * (1.0f / HID);
    out[(size_t)n * HID + c] = (__bf16)(d * rsqrtf(var + 1e-5f) * g[c] + b[c]);
}

// ---------------------------------------------------------------------------
// WMMA GEMM:  Y[M,OUT] = X[M,K] @ W[OUT,K]^T (+bias) (+relu) (+resid)
// bf16 inputs, f32 accumulate via v_wmma_f32_16x16x32_bf16.
// One wave computes a 16x64 output strip (4 accumulators): one A fragment
// feeds 4 WMMAs per K-step (4x A reuse). K is a compile-time constant so the
// K-loop fully unrolls -> register renaming -> loads hoisted across WMMAs
// (partial s_wait_loadcnt, XDL/memory overlap).
// EXEC stays all-ones (wave-uniform guard). Requires OUT % 64 == 0.
// ---------------------------------------------------------------------------
template <int K>
__global__ void k_gemm_wmma(const __bf16* __restrict__ X, const __bf16* __restrict__ W,
                            const float* __restrict__ bias, const float* __restrict__ resid,
                            float* __restrict__ outF, __bf16* __restrict__ outB,
                            int M, int OUT, int relu) {
    int wave    = (blockIdx.x * blockDim.x + threadIdx.x) >> 5;
    int lane    = threadIdx.x & 31;
    int tilesN  = OUT >> 6;                    // 64-wide strips
    int tileM   = wave / tilesN;
    int tn      = wave - tileM * tilesN;
    if (tileM * 16 >= M) return;               // wave-uniform

    int half = lane >> 4;                      // 0 | 1
    int l16  = lane & 15;

    // A (16x32 bf16): lane(l16) = row; lanes<16 hold K 0..15, lanes>=16 hold K 16..31
    const __bf16* pa = X + (size_t)(tileM * 16 + l16) * K + half * 16;
    // B (32x16 bf16): lane(l16) = col (row of W); lanes<16 hold K 0..15, lanes>=16 hold K 16..31
    const __bf16* pb0 = W + (size_t)(tn * 64 +  0 + l16) * K + half * 16;
    const __bf16* pb1 = W + (size_t)(tn * 64 + 16 + l16) * K + half * 16;
    const __bf16* pb2 = W + (size_t)(tn * 64 + 32 + l16) * K + half * 16;
    const __bf16* pb3 = W + (size_t)(tn * 64 + 48 + l16) * K + half * 16;

    v8f acc0 = {}, acc1 = {}, acc2 = {}, acc3 = {};
#pragma unroll
    for (int k0 = 0; k0 < K; k0 += 32) {
        v16bf a, b0, b1, b2, b3;
#pragma unroll
        for (int i = 0; i < 16; ++i) { a[i] = pa[k0 + i]; }
#pragma unroll
        for (int i = 0; i < 16; ++i) {
            b0[i] = pb0[k0 + i]; b1[i] = pb1[k0 + i];
            b2[i] = pb2[k0 + i]; b3[i] = pb3[k0 + i];
        }
        acc0 = __builtin_amdgcn_wmma_f32_16x16x32_bf16(false, a, false, b0, (short)0, acc0, false, false);
        acc1 = __builtin_amdgcn_wmma_f32_16x16x32_bf16(false, a, false, b1, (short)0, acc1, false, false);
        acc2 = __builtin_amdgcn_wmma_f32_16x16x32_bf16(false, a, false, b2, (short)0, acc2, false, false);
        acc3 = __builtin_amdgcn_wmma_f32_16x16x32_bf16(false, a, false, b3, (short)0, acc3, false, false);
    }

    // C/D layout: VGPR r -> row = r + 8*half, col = l16 (per 16x16 tile)
#pragma unroll
    for (int j = 0; j < 4; ++j) {
        v8f acc = (j == 0) ? acc0 : (j == 1) ? acc1 : (j == 2) ? acc2 : acc3;
        int col = tn * 64 + j * 16 + l16;
        float bval = bias ? bias[col] : 0.0f;
#pragma unroll
        for (int r = 0; r < 8; ++r) {
            int row = tileM * 16 + r + half * 8;
            size_t idx = (size_t)row * OUT + col;
            float v = acc[r] + bval;
            if (relu)  v = v > 0.0f ? v : 0.0f;
            if (resid) v += resid[idx];
            if (outF)  outF[idx] = v;
            if (outB)  outB[idx] = (__bf16)v;
        }
    }
}

// ---------------------------------------------------------------------------
// Edge pass 1: alpha[e,h] = <q[dst,h], k[src,h] + e_proj> / sqrt(C);  m = segmax
// ycat layout per node: [q(128) | k(128) | v(128) | skip(128)]
// ---------------------------------------------------------------------------
__global__ void k_edge_alpha(const int* __restrict__ ei, const float* __restrict__ ea,
                             const float* __restrict__ We, const float* __restrict__ ycat,
                             float* __restrict__ alpha, float* __restrict__ m, int E) {
    __shared__ float sWe[HID * 4];
    int t = threadIdx.x;
    sWe[t]       = We[t];
    sWe[t + 256] = We[t + 256];
    __syncthreads();

    int gid = blockIdx.x * 256 + t;
    if (gid >= E * NH) return;
    int e = gid >> 2, h = gid & 3;
    int src = ei[e], dst = ei[E + e];

    float a0 = ea[e * 4 + 0], a1 = ea[e * 4 + 1], a2 = ea[e * 4 + 2], a3 = ea[e * 4 + 3];
    const float* qrow = ycat + (size_t)dst * 512 + h * CH;
    const float* krow = ycat + (size_t)src * 512 + HID + h * CH;
    const float* w    = sWe + h * CH * 4;

    float s = 0.0f;
#pragma unroll
    for (int c = 0; c < CH; ++c) {
        float ep = a0 * w[c * 4] + a1 * w[c * 4 + 1] + a2 * w[c * 4 + 2] + a3 * w[c * 4 + 3];
        s += qrow[c] * (krow[c] + ep);
    }
    s *= 0.17677669529663687f; // 1/sqrt(32)
    alpha[gid] = s;
    atomicMaxF(&m[dst * NH + h], s);
}

// ---------------------------------------------------------------------------
// Edge pass 2: a = exp(alpha - m[dst]); denom += a; out[dst] += a * (v[src]+e_proj)
// ---------------------------------------------------------------------------
__global__ void k_edge_accum(const int* __restrict__ ei, const float* __restrict__ ea,
                             const float* __restrict__ We, const float* __restrict__ ycat,
                             const float* __restrict__ alpha, const float* __restrict__ m,
                             float* __restrict__ denom, float* __restrict__ outAcc, int E) {
    __shared__ float sWe[HID * 4];
    int t = threadIdx.x;
    sWe[t]       = We[t];
    sWe[t + 256] = We[t + 256];
    __syncthreads();

    int gid = blockIdx.x * 256 + t;
    if (gid >= E * NH) return;
    int e = gid >> 2, h = gid & 3;
    int src = ei[e], dst = ei[E + e];

    float aexp = __expf(alpha[gid] - m[dst * NH + h]);
    atomicAdd(&denom[dst * NH + h], aexp);

    float a0 = ea[e * 4 + 0], a1 = ea[e * 4 + 1], a2 = ea[e * 4 + 2], a3 = ea[e * 4 + 3];
    const float* vrow = ycat + (size_t)src * 512 + 2 * HID + h * CH;
    float*       orow = outAcc + (size_t)dst * HID + h * CH;
    const float* w    = sWe + h * CH * 4;
#pragma unroll
    for (int c = 0; c < CH; ++c) {
        float ep = a0 * w[c * 4] + a1 * w[c * 4 + 1] + a2 * w[c * 4 + 2] + a3 * w[c * 4 + 3];
        atomicAdd(&orow[c], aexp * (vrow[c] + ep));
    }
}

// ---------------------------------------------------------------------------
// Node combine: normalize attention, beta gate, residual, fused LN2 -> bf16
// one block of 128 threads per node
// ---------------------------------------------------------------------------
__global__ void k_node_combine(const float* __restrict__ x, const float* __restrict__ ycat,
                               const float* __restrict__ outAcc, const float* __restrict__ denom,
                               const float* __restrict__ Wbeta, const float* __restrict__ ln2g,
                               const float* __restrict__ ln2b, float* __restrict__ xmid,
                               __bf16* __restrict__ h2in) {
    __shared__ float sm[HID];
    int n = blockIdx.x, c = threadIdx.x;

    float den = denom[n * NH + (c >> 5)] + 1e-16f;
    float o   = outAcc[(size_t)n * HID + c] / den;
    float sk  = ycat[(size_t)n * 512 + 3 * HID + c];

    // beta = sigmoid([o | sk | o-sk] . Wbeta)
    float contrib = o * (Wbeta[c] + Wbeta[256 + c]) + sk * (Wbeta[128 + c] - Wbeta[256 + c]);
    float logit   = blk_reduce_sum128(contrib, sm);
    float beta    = 1.0f / (1.0f + __expf(-logit));

    float hmix = beta * sk + (1.0f - beta) * o;
    float xm   = x[(size_t)n * HID + c] + hmix;
    xmid[(size_t)n * HID + c] = xm;

    // fused LayerNorm2 -> bf16 for the FFN GEMM
    float mu  = blk_reduce_sum128(xm, sm) * (1.0f / HID);
    float d   = xm - mu;
    float var = blk_reduce_sum128(d * d, sm) * (1.0f / HID);
    h2in[(size_t)n * HID + c] = (__bf16)(d * rsqrtf(var + 1e-5f) * ln2g[c] + ln2b[c]);
}

// ---------------------------------------------------------------------------
// launch
// ---------------------------------------------------------------------------
extern "C" void kernel_launch(void* const* d_in, const int* in_sizes, int n_in,
                              void* d_out, int out_size, void* d_ws, size_t ws_size,
                              hipStream_t stream) {
    const int N = in_sizes[0] / HID;   // 50000
    const int E = in_sizes[2] / 4;     // 800000

    const float* x     = (const float*)d_in[0];
    const int*   ei    = (const int*)d_in[1];
    const float* ea    = (const float*)d_in[2];
    const float* ln1g  = (const float*)d_in[3];
    const float* ln1b  = (const float*)d_in[4];
    const float* Wq    = (const float*)d_in[5];
    const float* bq    = (const float*)d_in[6];
    const float* Wk    = (const float*)d_in[7];
    const float* bk    = (const float*)d_in[8];
    const float* Wv    = (const float*)d_in[9];
    const float* bv    = (const float*)d_in[10];
    const float* We    = (const float*)d_in[11];
    const float* Wsk   = (const float*)d_in[12];
    const float* bsk   = (const float*)d_in[13];
    const float* Wbeta = (const float*)d_in[14];
    const float* ln2g  = (const float*)d_in[15];
    const float* ln2b  = (const float*)d_in[16];
    const float* W1    = (const float*)d_in[17];
    const float* b1    = (const float*)d_in[18];
    const float* W2    = (const float*)d_in[19];
    const float* b2    = (const float*)d_in[20];

    auto al = [](size_t v) { return (v + 255) & ~(size_t)255; };
    char* base = (char*)d_ws;
    size_t off = 0;
    size_t o_ycat  = off; off += al((size_t)N * 512 * 4);  // q|k|v|skip f32; ffn1(bf16) aliases
    size_t o_out   = off; off += al((size_t)N * HID * 4);
    size_t o_xmid  = off; off += al((size_t)N * HID * 4);
    size_t o_alpha = off; off += al((size_t)E * NH * 4);   // h2in(bf16) aliases (same size)
    size_t o_m     = off; off += al((size_t)N * NH * 4);
    size_t o_den   = off; off += al((size_t)N * NH * 4);
    size_t o_h1b   = off; off += al((size_t)N * HID * 2);
    size_t o_wcat  = off; off += al((size_t)512 * HID * 2);
    size_t o_bcat  = off; off += al((size_t)512 * 4);
    size_t o_w1b   = off; off += al((size_t)512 * HID * 2);
    size_t o_w2b   = off; off += al((size_t)HID * 512 * 2);
    (void)off; (void)ws_size; (void)n_in; (void)out_size;

    float*  ycat   = (float*)(base + o_ycat);
    float*  outAcc = (float*)(base + o_out);
    float*  xmid   = (float*)(base + o_xmid);
    float*  alpha  = (float*)(base + o_alpha);
    float*  mbuf   = (float*)(base + o_m);
    float*  denom  = (float*)(base + o_den);
    __bf16* h1b    = (__bf16*)(base + o_h1b);
    __bf16* wcat   = (__bf16*)(base + o_wcat);
    float*  bcat   = (float*)(base + o_bcat);
    __bf16* w1b    = (__bf16*)(base + o_w1b);
    __bf16* w2b    = (__bf16*)(base + o_w2b);
    __bf16* ffn1b  = (__bf16*)(base + o_ycat);   // alias (ycat dead after node_combine)
    __bf16* h2in   = (__bf16*)(base + o_alpha);  // alias (alpha dead after edge_accum)
    float*  outF   = (float*)d_out;

    // 1. weight conversion f32 -> bf16 (Wcat = [q;k;v;skip] as [512,128])
    const int WN = HID * HID; // 16384
    k_f2bf<<<(WN + 255) / 256, 256, 0, stream>>>(Wq,  wcat + 0 * WN, WN);
    k_f2bf<<<(WN + 255) / 256, 256, 0, stream>>>(Wk,  wcat + 1 * WN, WN);
    k_f2bf<<<(WN + 255) / 256, 256, 0, stream>>>(Wv,  wcat + 2 * WN, WN);
    k_f2bf<<<(WN + 255) / 256, 256, 0, stream>>>(Wsk, wcat + 3 * WN, WN);
    k_f2bf<<<(512 * HID + 255) / 256, 256, 0, stream>>>(W1, w1b, 512 * HID);
    k_f2bf<<<(HID * 512 + 255) / 256, 256, 0, stream>>>(W2, w2b, HID * 512);
    hipMemcpyAsync(bcat + 0 * HID, bq,  HID * 4, hipMemcpyDeviceToDevice, stream);
    hipMemcpyAsync(bcat + 1 * HID, bk,  HID * 4, hipMemcpyDeviceToDevice, stream);
    hipMemcpyAsync(bcat + 2 * HID, bv,  HID * 4, hipMemcpyDeviceToDevice, stream);
    hipMemcpyAsync(bcat + 3 * HID, bsk, HID * 4, hipMemcpyDeviceToDevice, stream);

    // 2. LN1 -> bf16 ; init accumulators
    k_ln_bf16<<<N, HID, 0, stream>>>(x, ln1g, ln1b, h1b);
    k_init<<<(N * HID + 255) / 256, 256, 0, stream>>>(outAcc, (unsigned int*)mbuf, denom, N);

    // 3. fused QKV+skip GEMM: ycat[N,512] = h1b @ Wcat^T + bcat  (16x64 per wave)
    {
        int tiles = (N / 16) * (512 / 64);
        k_gemm_wmma<HID><<<(tiles + 7) / 8, 256, 0, stream>>>(h1b, wcat, bcat, nullptr,
                                                              ycat, nullptr, N, 512, 0);
    }

    // 4. edge phase
    int eblocks = (E * NH + 255) / 256;
    k_edge_alpha<<<eblocks, 256, 0, stream>>>(ei, ea, We, ycat, alpha, mbuf, E);
    k_edge_accum<<<eblocks, 256, 0, stream>>>(ei, ea, We, ycat, alpha, mbuf, denom, outAcc, E);

    // 5. node combine (+ fused LN2 -> bf16)
    k_node_combine<<<N, HID, 0, stream>>>(x, ycat, outAcc, denom, Wbeta, ln2g, ln2b, xmid, h2in);

    // 6. FFN GEMM1: relu(h2in @ W1^T + b1) -> bf16 [N,512]
    {
        int tiles = (N / 16) * (512 / 64);
        k_gemm_wmma<HID><<<(tiles + 7) / 8, 256, 0, stream>>>(h2in, w1b, b1, nullptr,
                                                              nullptr, ffn1b, N, 512, 1);
    }
    // 7. FFN GEMM2 + residual: d_out = xmid + ffn1 @ W2^T + b2
    {
        int tiles = (N / 16) * (HID / 64);
        k_gemm_wmma<512><<<(tiles + 7) / 8, 256, 0, stream>>>(ffn1b, w2b, b2, xmid,
                                                              outF, nullptr, N, HID, 0);
    }
}
